// Seq2Seq_a_59949153517894
// MI455X (gfx1250) — matrix operand order repaired and checked by reading, not compiled, not run on hardware
//
#include <hip/hip_runtime.h>
#include <hip/hip_bf16.h>

// ---------------- types ----------------
typedef __attribute__((ext_vector_type(16))) _Float16 v16h;
typedef __attribute__((ext_vector_type(8)))  _Float16 v8h;
typedef __attribute__((ext_vector_type(8)))  float    v8f;

#define B_    128
#define S_    512
#define T_    200
#define E_    256
#define H_    512
#define G4_   2048     // 4*H
#define KEH_  768      // E + H
#define DECV_ 200
#define NPAD_ 208      // DECV padded to multiple of 16
#define BH_   (B_ * H_)

__device__ __forceinline__ float sigmoidf_(float x) {
  return 1.0f / (1.0f + __expf(-x));
}

// ---- WMMA fragment loaders (16x16x32 f16, wave32 layouts per CDNA5 ISA) ----
// A (16x32): lanes 0-15 row M=lane, K = k0+0..7 (v0-3) and k0+16..23 (v4-7);
//            lanes 16-31 row M=lane-16, K offsets +8 / +24.
__device__ __forceinline__ v16h load_a_frag(const _Float16* base, int ld, int m0, int k0) {
  const int lane = (int)(threadIdx.x & 31u);
  const int row  = m0 + (lane & 15);
  const int kof  = k0 + ((lane >> 4) << 3);
  const _Float16* p = base + (size_t)row * ld + kof;
  v8h lo = *(const v8h*)(p);
  v8h hi = *(const v8h*)(p + 16);
  return __builtin_shufflevector(lo, hi, 0,1,2,3,4,5,6,7,8,9,10,11,12,13,14,15);
}
// B (32x16, row-major K x N): lane L holds col N=n0+(L&15), K = k0+0..7 and k0+16..23
//            (lanes 16-31: K offsets +8 / +24), matching the A fragment.
__device__ __forceinline__ v16h load_b_frag(const _Float16* base, int ld, int k0, int n0) {
  const int lane = (int)(threadIdx.x & 31u);
  const _Float16* p = base + (size_t)(k0 + ((lane >> 4) << 3)) * ld + n0 + (lane & 15);
  v16h r;
#pragma unroll
  for (int s = 0; s < 8; ++s) { r[s] = p[(size_t)s * ld]; r[s + 8] = p[(size_t)(s + 16) * ld]; }
  return r;
}

// ---------------- one LSTM time-step (512 threads = 16 waves) ----------------
// Gates g[B][4H] = X[B][E] @ Bmat[0:256,:] + h[B][H] @ Bmat[256:768,:] + bias,
// then pointwise update. Each wave owns 16 "quadruples" (mt, jt): output tiles
// (mt, jt), (mt, jt+32), (mt, jt+64), (mt, jt+96) == gates i,f,g,o for the same
// hidden column block -> accumulators are element-aligned for the pointwise.
__device__ __forceinline__ void lstm_step(
    const _Float16* __restrict__ Xs,    // [B][E] this step (global, f16)
    const _Float16* __restrict__ hcur,  // [B][H] (LDS, f16)
    _Float16*       __restrict__ hnext, // [B][H] (LDS, f16)
    const _Float16* __restrict__ Bmat,  // [768][2048] f16
    const float*    __restrict__ bias,  // [2048] = bih + bhh
    float*          __restrict__ cws,   // [B][H] f32 running cell state
    bool czero,                         // first step: c==0 and h==0 (skip h GEMM)
    float* __restrict__ hout,           // optional final-h f32 sink (may be null)
    float* __restrict__ cout)           // optional final-c f32 sink (may be null)
{
  const int wave  = (int)(threadIdx.x >> 5);
  const int lane  = (int)(threadIdx.x & 31u);
  const int bhalf = (lane >> 4) << 3;   // 0 or 8
  const int ncol  = lane & 15;

#pragma unroll 1
  for (int q = wave * 16; q < wave * 16 + 16; ++q) {
    const int mt = q >> 5;          // 0..7   (batch tile)
    const int jt = q & 31;          // 0..31  (hidden column tile)
    const int m0 = mt << 4;
    const int j0 = jt << 4;
    v8f ai = {}, af4 = {}, ag = {}, ao = {};

    // x contribution: K = 0..255
#pragma unroll 2
    for (int kc = 0; kc < E_; kc += 32) {
      v16h a  = load_a_frag(Xs, E_, m0, kc);
      v16h b0 = load_b_frag(Bmat, G4_, kc, j0);
      v16h b1 = load_b_frag(Bmat, G4_, kc, j0 + H_);
      v16h b2 = load_b_frag(Bmat, G4_, kc, j0 + 2 * H_);
      v16h b3 = load_b_frag(Bmat, G4_, kc, j0 + 3 * H_);
      ai  = __builtin_amdgcn_wmma_f32_16x16x32_f16(false, a, false, b0, (short)0, ai,  false, false);
      af4 = __builtin_amdgcn_wmma_f32_16x16x32_f16(false, a, false, b1, (short)0, af4, false, false);
      ag  = __builtin_amdgcn_wmma_f32_16x16x32_f16(false, a, false, b2, (short)0, ag,  false, false);
      ao  = __builtin_amdgcn_wmma_f32_16x16x32_f16(false, a, false, b3, (short)0, ao,  false, false);
    }
    // h contribution: K = 256..767 (skipped on the very first step, h==0)
    if (!czero) {
      const _Float16* Bh = Bmat + (size_t)E_ * G4_;
#pragma unroll 2
      for (int kc = 0; kc < H_; kc += 32) {
        v16h a  = load_a_frag(hcur, H_, m0, kc);
        v16h b0 = load_b_frag(Bh, G4_, kc, j0);
        v16h b1 = load_b_frag(Bh, G4_, kc, j0 + H_);
        v16h b2 = load_b_frag(Bh, G4_, kc, j0 + 2 * H_);
        v16h b3 = load_b_frag(Bh, G4_, kc, j0 + 3 * H_);
        ai  = __builtin_amdgcn_wmma_f32_16x16x32_f16(false, a, false, b0, (short)0, ai,  false, false);
        af4 = __builtin_amdgcn_wmma_f32_16x16x32_f16(false, a, false, b1, (short)0, af4, false, false);
        ag  = __builtin_amdgcn_wmma_f32_16x16x32_f16(false, a, false, b2, (short)0, ag,  false, false);
        ao  = __builtin_amdgcn_wmma_f32_16x16x32_f16(false, a, false, b3, (short)0, ao,  false, false);
      }
    }

    // pointwise LSTM (gate order i,f,g,o per PyTorch / reference split)
    const int   j  = j0 + ncol;
    const float bi = bias[j];
    const float bf = bias[j + H_];
    const float bg = bias[j + 2 * H_];
    const float bo = bias[j + 3 * H_];
    union U { v8f v; float f[8]; };
    U ui, uf, ug, uo;
    ui.v = ai; uf.v = af4; ug.v = ag; uo.v = ao;
#pragma unroll
    for (int r = 0; r < 8; ++r) {
      const int b  = m0 + bhalf + r;        // C/D layout: VGPR r, half-wave -> row
      float cold   = czero ? 0.0f : cws[b * H_ + j];
      float iv = sigmoidf_(ui.f[r] + bi);
      float fv = sigmoidf_(uf.f[r] + bf);
      float gv = tanhf(ug.f[r] + bg);
      float ov = sigmoidf_(uo.f[r] + bo);
      float cn = fv * cold + iv * gv;
      float hv = ov * tanhf(cn);
      cws[b * H_ + j]   = cn;
      hnext[b * H_ + j] = (_Float16)hv;
      if (hout) { hout[b * H_ + j] = hv; cout[b * H_ + j] = cn; }
    }
  }
}

// ---------------- weight / input prep kernels ----------------
// Combined gate B matrix [768][2048] f16: rows 0..255 = Wih^T, 256..767 = Whh^T.
__global__ void prep_gateB(const float* __restrict__ Wih, const float* __restrict__ Whh,
                           const float* __restrict__ bih, const float* __restrict__ bhh,
                           _Float16* __restrict__ Bm, float* __restrict__ bias) {
  for (int idx = blockIdx.x * blockDim.x + threadIdx.x; idx < KEH_ * G4_;
       idx += gridDim.x * blockDim.x) {
    int k = idx / G4_, n = idx % G4_;
    float v = (k < E_) ? Wih[(size_t)n * E_ + k] : Whh[(size_t)n * H_ + (k - E_)];
    Bm[idx] = (_Float16)v;
  }
  for (int n = blockIdx.x * blockDim.x + threadIdx.x; n < G4_;
       n += gridDim.x * blockDim.x)
    bias[n] = bih[n] + bhh[n];
}

__global__ void prep_woutT(const float* __restrict__ Wout, _Float16* __restrict__ WT) {
  for (int idx = blockIdx.x * blockDim.x + threadIdx.x; idx < H_ * NPAD_;
       idx += gridDim.x * blockDim.x) {
    int k = idx / NPAD_, n = idx % NPAD_;
    WT[idx] = (n < DECV_) ? (_Float16)Wout[(size_t)n * H_ + k] : (_Float16)0.0f;
  }
}

__global__ void gather_src(const int* __restrict__ seq, const float* __restrict__ emb,
                           _Float16* __restrict__ X) {
  for (int idx = blockIdx.x * blockDim.x + threadIdx.x; idx < S_ * B_ * E_;
       idx += gridDim.x * blockDim.x) {
    int s = idx / (B_ * E_), r = idx % (B_ * E_), b = r / E_, e = r % E_;
    int tok = seq[b * S_ + s];
    X[idx] = (_Float16)emb[(size_t)tok * E_ + e];
  }
}

__global__ void gather_dec(const int* __restrict__ seq, const float* __restrict__ emb,
                           _Float16* __restrict__ X) {
  for (int idx = blockIdx.x * blockDim.x + threadIdx.x; idx < (T_ - 1) * B_ * E_;
       idx += gridDim.x * blockDim.x) {
    int t = idx / (B_ * E_), r = idx % (B_ * E_), b = r / E_, e = r % E_;
    int tok = seq[b * T_ + t];                 // target_seq[:, :-1]
    X[idx] = (_Float16)emb[(size_t)tok * E_ + e];
  }
}

// ---------------- encoder: 2 blocks (fwd / rev), 512 threads each ----------------
__global__ void __launch_bounds__(512)
encoder_lstm(const _Float16* __restrict__ Xsrc,   // [S][B][E]
             const _Float16* __restrict__ Bmat_f, const _Float16* __restrict__ Bmat_r,
             const float* __restrict__ bias_f,    const float* __restrict__ bias_r,
             float* __restrict__ c_ws,            // [2][B][H]
             float* __restrict__ hfin, float* __restrict__ cfin) { // [2][B][H]
  extern __shared__ char smem_raw[];
  _Float16* hbuf = (_Float16*)smem_raw;            // double buffer [2][B][H]
  const int dir = blockIdx.x;
  const _Float16* Bmat = dir ? Bmat_r : Bmat_f;
  const float* bias    = dir ? bias_r : bias_f;
  float* cws           = c_ws + (size_t)dir * BH_;

  for (int si = 0; si < S_; ++si) {
    const int s = dir ? (S_ - 1 - si) : si;
    const bool last = (si == S_ - 1);
    lstm_step(Xsrc + (size_t)s * (B_ * E_),
              hbuf + (size_t)(si & 1) * BH_,
              hbuf + (size_t)((si + 1) & 1) * BH_,
              Bmat, bias, cws, si == 0,
              last ? hfin + (size_t)dir * BH_ : nullptr,
              last ? cfin + (size_t)dir * BH_ : nullptr);
    __syncthreads();
  }
}

// ---------------- bridge: h0/c0 = cat(h_f,h_r)/cat(c_f,c_r) @ W^T + b ----------------
__global__ void bridge_kernel(const float* __restrict__ hfin, const float* __restrict__ cfin,
                              const float* __restrict__ Whid, const float* __restrict__ bhid,
                              const float* __restrict__ Wcell, const float* __restrict__ bcell,
                              _Float16* __restrict__ h0f16, float* __restrict__ cdec) {
  int idx = blockIdx.x * blockDim.x + threadIdx.x;
  if (idx >= BH_) return;
  int b = idx / H_, j = idx % H_;
  const float* hf = hfin + (size_t)b * H_;
  const float* hr = hfin + (size_t)BH_ + (size_t)b * H_;
  const float* cf = cfin + (size_t)b * H_;
  const float* cr = cfin + (size_t)BH_ + (size_t)b * H_;
  const float* wh = Whid + (size_t)j * 2 * H_;
  const float* wc = Wcell + (size_t)j * 2 * H_;
  float ah = bhid[j], ac = bcell[j];
  for (int k = 0; k < H_; ++k) { ah += hf[k] * wh[k];       ac += cf[k] * wc[k]; }
  for (int k = 0; k < H_; ++k) { ah += hr[k] * wh[H_ + k];  ac += cr[k] * wc[H_ + k]; }
  h0f16[idx] = (_Float16)ah;
  cdec[idx]  = ac;
}

// ---------------- decoder: 1 block, 512 threads, 199 steps ----------------
__global__ void __launch_bounds__(512)
decoder_lstm(const _Float16* __restrict__ Xdec,   // [T-1][B][E]
             const _Float16* __restrict__ Bmat,   // [768][2048]
             const float* __restrict__ bias,
             const _Float16* __restrict__ h0f16,  // [B][H]
             float* __restrict__ cdec,            // [B][H] preset to c0
             const _Float16* __restrict__ WoutT,  // [H][NPAD]
             const float* __restrict__ bout,      // [200]
             const int* __restrict__ target_seq,  // [B][T]
             float* __restrict__ out,             // [B][T][V]
             float* __restrict__ pred) {          // [B][T]
  extern __shared__ char smem_raw[];
  _Float16* hbuf = (_Float16*)smem_raw;           // [2][B][H]
  const int wave  = (int)(threadIdx.x >> 5);
  const int lane  = (int)(threadIdx.x & 31u);
  const int bhalf = (lane >> 4) << 3;
  const int ncol  = lane & 15;

  // init: h0 -> LDS, zero t=0 logits row, seed pred[:,0]
  for (int i = threadIdx.x; i < BH_; i += blockDim.x) hbuf[i] = h0f16[i];
  for (int i = threadIdx.x; i < B_ * DECV_; i += blockDim.x)
    out[(size_t)(i / DECV_) * T_ * DECV_ + (i % DECV_)] = 0.0f;
  if (threadIdx.x < B_)
    pred[(size_t)threadIdx.x * T_] = (float)target_seq[(size_t)threadIdx.x * T_];
  __syncthreads();

  for (int t = 0; t < T_ - 1; ++t) {
    const _Float16* hnext = hbuf + (size_t)((t + 1) & 1) * BH_;
    lstm_step(Xdec + (size_t)t * (B_ * E_),
              hbuf + (size_t)(t & 1) * BH_,
              (_Float16*)hnext, Bmat, bias, cdec, false, nullptr, nullptr);
    __syncthreads();

    // logits = h @ WoutT + bout : M=128, K=512, N=208(pad) -> 8 x 13 tiles
#pragma unroll 1
    for (int tt = wave; tt < 8 * 13; tt += 16) {
      const int mt = tt / 13, nt = tt % 13;
      const int m0 = mt << 4, n0 = nt << 4;
      v8f acc = {};
#pragma unroll 2
      for (int kc = 0; kc < H_; kc += 32) {
        v16h a = load_a_frag(hnext, H_, m0, kc);
        v16h b = load_b_frag(WoutT, NPAD_, kc, n0);
        acc = __builtin_amdgcn_wmma_f32_16x16x32_f16(false, a, false, b, (short)0, acc, false, false);
      }
      const int col = n0 + ncol;
      if (col < DECV_) {
        const float bb = bout[col];
        union U { v8f v; float f[8]; } u; u.v = acc;
#pragma unroll
        for (int r = 0; r < 8; ++r) {
          const int b = m0 + bhalf + r;
          out[((size_t)b * T_ + (t + 1)) * DECV_ + col] = u.f[r] + bb;
        }
      }
    }
    __threadfence_block();
    __syncthreads();

    // argmax over vocab (first max index, matching jnp.argmax)
    if (threadIdx.x < B_) {
      const int b = threadIdx.x;
      const float* row = out + ((size_t)b * T_ + (t + 1)) * DECV_;
      float best = row[0]; int bidx = 0;
      for (int v = 1; v < DECV_; ++v) { float x = row[v]; if (x > best) { best = x; bidx = v; } }
      pred[(size_t)b * T_ + (t + 1)] = (float)bidx;
    }
    __syncthreads();
  }
}

// ---------------- host launcher ----------------
extern "C" void kernel_launch(void* const* d_in, const int* in_sizes, int n_in,
                              void* d_out, int out_size, void* d_ws, size_t ws_size,
                              hipStream_t stream) {
  (void)in_sizes; (void)n_in; (void)out_size; (void)ws_size;
  const int*   source_seq = (const int*)  d_in[0];
  const int*   target_seq = (const int*)  d_in[1];
  const float* embed      = (const float*)d_in[2];
  const float* eWih_f = (const float*)d_in[3];  const float* eWhh_f = (const float*)d_in[4];
  const float* ebih_f = (const float*)d_in[5];  const float* ebhh_f = (const float*)d_in[6];
  const float* eWih_r = (const float*)d_in[7];  const float* eWhh_r = (const float*)d_in[8];
  const float* ebih_r = (const float*)d_in[9];  const float* ebhh_r = (const float*)d_in[10];
  const float* Whid  = (const float*)d_in[11];  const float* bhid  = (const float*)d_in[12];
  const float* Wcell = (const float*)d_in[13];  const float* bcell = (const float*)d_in[14];
  const float* dec_emb = (const float*)d_in[15];
  const float* dWih = (const float*)d_in[16];   const float* dWhh = (const float*)d_in[17];
  const float* dbih = (const float*)d_in[18];   const float* dbhh = (const float*)d_in[19];
  const float* Wout = (const float*)d_in[20];   const float* bout = (const float*)d_in[21];

  float* out  = (float*)d_out;
  float* pred = out + (size_t)B_ * T_ * DECV_;

  // workspace carve-up (256B aligned)
  char* w = (char*)d_ws;
  size_t off = 0;
  auto take = [&](size_t bytes) -> char* {
    char* p = w + off;
    off = (off + bytes + 255) & ~(size_t)255;
    return p;
  };
  _Float16* Benc_f = (_Float16*)take((size_t)KEH_ * G4_ * 2);
  _Float16* Benc_r = (_Float16*)take((size_t)KEH_ * G4_ * 2);
  _Float16* Bdec   = (_Float16*)take((size_t)KEH_ * G4_ * 2);
  _Float16* WoutT  = (_Float16*)take((size_t)H_ * NPAD_ * 2);
  float* bias_f = (float*)take((size_t)G4_ * 4);
  float* bias_r = (float*)take((size_t)G4_ * 4);
  float* bias_d = (float*)take((size_t)G4_ * 4);
  _Float16* Xsrc = (_Float16*)take((size_t)S_ * B_ * E_ * 2);
  _Float16* Xdec = (_Float16*)take((size_t)(T_ - 1) * B_ * E_ * 2);
  float* c_enc = (float*)take((size_t)2 * BH_ * 4);
  float* hfin  = (float*)take((size_t)2 * BH_ * 4);
  float* cfin  = (float*)take((size_t)2 * BH_ * 4);
  float* cdec  = (float*)take((size_t)BH_ * 4);
  _Float16* h0f16 = (_Float16*)take((size_t)BH_ * 2);

  // 1) weight / embedding prep
  prep_gateB<<<1024, 256, 0, stream>>>(eWih_f, eWhh_f, ebih_f, ebhh_f, Benc_f, bias_f);
  prep_gateB<<<1024, 256, 0, stream>>>(eWih_r, eWhh_r, ebih_r, ebhh_r, Benc_r, bias_r);
  prep_gateB<<<1024, 256, 0, stream>>>(dWih,   dWhh,   dbih,   dbhh,   Bdec,   bias_d);
  prep_woutT<<<256, 256, 0, stream>>>(Wout, WoutT);
  gather_src<<<4096, 256, 0, stream>>>(source_seq, embed, Xsrc);
  gather_dec<<<2048, 256, 0, stream>>>(target_seq, dec_emb, Xdec);

  // 2) bidirectional encoder (two independent persistent workgroups)
  const size_t lds = (size_t)2 * BH_ * sizeof(_Float16); // 256 KB double-buffered h
  encoder_lstm<<<2, 512, lds, stream>>>(Xsrc, Benc_f, Benc_r, bias_f, bias_r,
                                        c_enc, hfin, cfin);

  // 3) bridge
  bridge_kernel<<<(BH_ + 255) / 256, 256, 0, stream>>>(hfin, cfin, Whid, bhid,
                                                       Wcell, bcell, h0f16, cdec);

  // 4) decoder + logits + argmax
  decoder_lstm<<<1, 512, lds, stream>>>(Xdec, Bdec, bias_d, h0f16, cdec,
                                        WoutT, bout, target_seq, out, pred);
}
